// BandSplitRoFormerBackbone_61641370632203
// MI455X (gfx1250) — matrix operand order repaired and checked by reading, not compiled, not run on hardware
//
#include <hip/hip_runtime.h>
#include <hip/hip_bf16.h>

// ---------------------------------------------------------------------------
// BandSplit RoFormer backbone for gfx1250 (MI455X), wave32 + WMMA f16 + TDM.
// ---------------------------------------------------------------------------

typedef _Float16 f16;
typedef __attribute__((ext_vector_type(2)))  _Float16 v2h;
typedef __attribute__((ext_vector_type(8)))  _Float16 v8h;
typedef __attribute__((ext_vector_type(16))) _Float16 v16h;
typedef __attribute__((ext_vector_type(8)))  float    v8f;
typedef __attribute__((ext_vector_type(4)))  unsigned int u32x4;
typedef __attribute__((ext_vector_type(8)))  int i32x8;
typedef __attribute__((ext_vector_type(4)))  int i32x4;

#define DD     384
#define FFC    1536
#define NHEADC 8
#define HDC    48
#define NBLK   6
#define NTOK   31744   // 124*256 == 512*62

// Tensor Data Mover availability (this toolchain: 6-arg builtin).
#if defined(__has_builtin)
# if __has_builtin(__builtin_amdgcn_tensor_load_to_lds) && \
     __has_builtin(__builtin_amdgcn_s_wait_tensorcnt)
#  define HAVE_TDM 1
# endif
#endif
#ifndef HAVE_TDM
# define HAVE_TDM 0
#endif

// Strided token addressing: off = base + (seq/div)*hi + (seq%div)*lo + pos*ps
struct Layout {
  long base, hi_stride, lo_stride, pos_stride;
  int  seq_div;
};

__device__ __forceinline__ long tok_off(const Layout& L, int seq, int pos) {
  return L.base + (long)(seq / L.seq_div) * L.hi_stride
                + (long)(seq % L.seq_div) * L.lo_stride
                + (long)pos * L.pos_stride;
}

__device__ __forceinline__ v8f wmma16(v16h a, v16h b, v8f c) {
  // D = A(16x32 f16) x B(32x16 f16) + C(16x16 f32)
  return __builtin_amdgcn_wmma_f32_16x16x32_f16(false, a, false, b,
                                                (short)0, c, false, false);
}

// A-matrix fragment (16x32 f16, MxK), ISA 7.12.2 layout.
// ld must be a multiple of 8 halves.
__device__ __forceinline__ v16h load_a_frag(const f16* base, int ld, int koff) {
  int lane = threadIdx.x & 31;
  int m    = lane & 15;
  int kb   = ((lane >> 4) << 3) + koff;
  const f16* p = base + m * ld + kb;
  v8h lo = *(const v8h*)(p);
  v8h hi = *(const v8h*)(p + 16);
  v16h r;
#pragma unroll
  for (int i = 0; i < 8; ++i) { r[i] = lo[i]; r[i + 8] = hi[i]; }
  return r;
}

// B-matrix fragment (32x16 f16, KxN) where B[k][n] = src[n*ld + k].
__device__ __forceinline__ v16h load_b_fragT(const f16* base, int ld, int koff) {
  int lane = threadIdx.x & 31;
  int n    = lane & 15;
  int kb   = ((lane >> 4) << 3) + koff;
  const f16* p = base + n * ld + kb;
  v8h lo = *(const v8h*)(p);
  v8h hi = *(const v8h*)(p + 16);
  v16h r;
#pragma unroll
  for (int i = 0; i < 8; ++i) { r[i] = lo[i]; r[i + 8] = hi[i]; }
  return r;
}

__device__ __forceinline__ float gelu_exact(float v) {
  return 0.5f * v * (1.0f + erff(v * 0.70710678118654752f));
}

#if HAVE_TDM
// LDS byte offset of a pointer into static shared memory.
__device__ __forceinline__ unsigned lds_offset_of(const void* p) {
  return (unsigned)(unsigned long long)
         (__attribute__((address_space(3))) const void*)p;
}

// TDM 2D tile load: tile_h rows x tile_w f16 elements from a row-major
// tensor (row stride row_stride elements) into LDS at lds_off, inserting
// 4 DWORDs (8 halves) of padding after every 16 DWORDs (32 halves) stored,
// i.e. LDS row pitch = (tile_w/32)*40 halves.
__device__ __forceinline__ void tdm_load_2d(const void* gaddr, unsigned lds_off,
                                            unsigned tensor_w, unsigned tensor_h,
                                            unsigned long long row_stride,
                                            unsigned tile_w, unsigned tile_h) {
  unsigned long long ga = (unsigned long long)gaddr;
  u32x4 g0;
  g0.x = 1u;                                        // count=1, user descriptor
  g0.y = lds_off;                                   // LDS byte address
  g0.z = (unsigned)(ga & 0xffffffffull);            // global_addr[31:0]
  g0.w = (unsigned)((ga >> 32) & 0x1ffffffull)      // global_addr[56:32]
       | (2u << 30);                                // type = 2 ("image")
  i32x8 g1;
  g1[0] = (int)((1u << 16)                          // data_size = 1 (2 bytes)
              | (1u << 20)                          // pad_enable
              | (3u << 22)                          // pad_interval: 16 DWORDs
              | (3u << 25));                        // pad_amount: 4 DWORDs
  g1[1] = (int)((tensor_w & 0xffffu) << 16);        // dim0[15:0] (barrier addr 0)
  g1[2] = (int)(((tensor_w >> 16) & 0xffffu)        // dim0[31:16]
              | ((tensor_h & 0xffffu) << 16));      // dim1[15:0]
  g1[3] = (int)(((tensor_h >> 16) & 0xffffu)        // dim1[31:16]
              | ((tile_w & 0xffffu) << 16));        // tile_dim0
  g1[4] = (int)(tile_h & 0xffffu);                  // tile_dim1 (tile_dim2 = 0)
  g1[5] = (int)(row_stride & 0xffffffffull);        // dim0_stride[31:0]
  g1[6] = (int)((row_stride >> 32) & 0xffffull);    // dim0_stride[47:32]
  g1[7] = 0;
  i32x4 z4 = {0, 0, 0, 0};
  i32x8 z8 = {0, 0, 0, 0, 0, 0, 0, 0};
  __builtin_amdgcn_tensor_load_to_lds(g0, g1, z4, z4, z8, 0);
}
#endif

// ---------------------------------------------------------------------------
// f32 -> f16 weight conversion
// ---------------------------------------------------------------------------
__global__ void cvt_f32_f16(const float* __restrict__ in, f16* __restrict__ out, long n) {
  long i = (long)blockIdx.x * blockDim.x + threadIdx.x;
  long stride = (long)gridDim.x * blockDim.x;
  for (; i < n; i += stride) out[i] = (f16)in[i];
}

// ---------------------------------------------------------------------------
// Fused RMSNorm (+ optional RoPE). 192 threads per token, one pair each.
// ---------------------------------------------------------------------------
__global__ void __launch_bounds__(192)
rmsnorm_rope_kernel(const float* __restrict__ x, const float* __restrict__ w,
                    f16* __restrict__ hout, f16* __restrict__ hrout,
                    Layout L, int Tlen) {
  int tok = blockIdx.x;
  int seq = tok / Tlen, pos = tok % Tlen;
  long off = tok_off(L, seq, pos);
  int i = threadIdx.x;                 // 0..191 (pair index)
  float x0 = x[off + 2 * i];
  float x1 = x[off + 2 * i + 1];
  float ss = x0 * x0 + x1 * x1;
#pragma unroll
  for (int m = 16; m >= 1; m >>= 1) ss += __shfl_xor(ss, m);
  __shared__ float red[6];
  if ((i & 31) == 0) red[i >> 5] = ss;
  __syncthreads();
  float tot = red[0] + red[1] + red[2] + red[3] + red[4] + red[5];
  float rs = rsqrtf(tot * (1.0f / (float)DD) + 1e-5f);
  float h0 = x0 * rs * w[2 * i];
  float h1 = x1 * rs * w[2 * i + 1];
  v2h hp; hp[0] = (f16)h0; hp[1] = (f16)h1;
  *(v2h*)(hout + (long)tok * DD + 2 * i) = hp;
  if (hrout) {
    float invf = __expf(-((float)(2 * i) * (1.0f / (float)DD)) * 9.210340371976184f);
    float ang = (float)pos * invf;
    float c = __cosf(ang), s = __sinf(ang);
    v2h rp; rp[0] = (f16)(h0 * c - h1 * s); rp[1] = (f16)(h0 * s + h1 * c);
    *(v2h*)(hrout + (long)tok * DD + 2 * i) = rp;
  }
}

// ---------------------------------------------------------------------------
// GEMM core: acc(2 x 16x16 tiles per wave) over C[M,N] = A[M,K] @ W[N,K]^T.
// K is consumed in double-buffered 128-wide chunks; each chunk tile is
// 64 rows x 128 halves stored with a 40-half-per-32 padded pitch (LDC=160).
// TDM path: wave 0 issues two tensor_load_to_lds per chunk and pipelines
// with s_wait_tensorcnt; fallback: cooperative b128 staging.
// ---------------------------------------------------------------------------
#define KC  128
#define LDC 160   // (KC/32)*40 halves per LDS row

__device__ __forceinline__ void gemm_core(const f16* __restrict__ A,
                                          const f16* __restrict__ W,
                                          int K, int tm0, int tn0,
                                          f16* Asb, f16* Wsb,   // [2][64*LDC]
                                          v8f& acc0, v8f& acc1) {
  const int tid  = threadIdx.x;
  const int wave = tid >> 5;
  const int wm   = wave >> 1;        // 0..3
  const int wn   = (wave & 1) * 2;   // 0 or 2
  const int nchunk = K / KC;
#if HAVE_TDM
  if (wave == 0) {
    tdm_load_2d(&A[(long)tm0 * K], lds_offset_of(Asb), K, 64, K, KC, 64);
    tdm_load_2d(&W[(long)tn0 * K], lds_offset_of(Wsb), K, 64, K, KC, 64);
  }
#else
  const int lrow = tid >> 2;         // 0..63
  const int lcol = (tid & 3) * 8;    // 0,8,16,24
#endif
  for (int c = 0; c < nchunk; ++c) {
    f16* Ab = Asb + (c & 1) * (64 * LDC);
    f16* Wb = Wsb + (c & 1) * (64 * LDC);
#if HAVE_TDM
    if (wave == 0) {
      if (c + 1 < nchunk) {
        f16* An = Asb + ((c + 1) & 1) * (64 * LDC);
        f16* Wn = Wsb + ((c + 1) & 1) * (64 * LDC);
        tdm_load_2d(&A[(long)tm0 * K + (c + 1) * KC], lds_offset_of(An),
                    K, 64, K, KC, 64);
        tdm_load_2d(&W[(long)tn0 * K + (c + 1) * KC], lds_offset_of(Wn),
                    K, 64, K, KC, 64);
        __builtin_amdgcn_s_wait_tensorcnt(2);  // chunk c complete, c+1 in flight
      } else {
        __builtin_amdgcn_s_wait_tensorcnt(0);
      }
    }
#else
#pragma unroll
    for (int kk = 0; kk < KC / 32; ++kk) {
      *(v8h*)&Ab[lrow * LDC + kk * 40 + lcol] =
          *(const v8h*)&A[(long)(tm0 + lrow) * K + c * KC + kk * 32 + lcol];
      *(v8h*)&Wb[lrow * LDC + kk * 40 + lcol] =
          *(const v8h*)&W[(long)(tn0 + lrow) * K + c * KC + kk * 32 + lcol];
    }
    if (c + 1 < nchunk) {
      __builtin_prefetch(&A[(long)(tm0 + lrow) * K + (c + 1) * KC + lcol], 0, 3);
      __builtin_prefetch(&W[(long)(tn0 + lrow) * K + (c + 1) * KC + lcol], 0, 3);
    }
#endif
    __syncthreads();
#pragma unroll
    for (int kk = 0; kk < KC / 32; ++kk) {
      v16h a  = load_a_frag(&Ab[(wm * 16) * LDC + kk * 40], LDC, 0);
      v16h b0 = load_b_fragT(&Wb[(wn * 16) * LDC + kk * 40], LDC, 0);
      v16h b1 = load_b_fragT(&Wb[((wn + 1) * 16) * LDC + kk * 40], LDC, 0);
      acc0 = wmma16(a, b0, acc0);
      acc1 = wmma16(a, b1, acc1);
    }
    __syncthreads();
  }
}

// C[M,N] = act(A @ W^T) in f16.  ACT: 0 = identity, 1 = exact GELU.
template <int ACT>
__global__ void __launch_bounds__(256)
gemm_tn_f16(const f16* __restrict__ A, const f16* __restrict__ W,
            f16* __restrict__ C, int M, int N, int K) {
  __shared__ f16 Asb[2][64 * LDC];
  __shared__ f16 Wsb[2][64 * LDC];
  const int tn0 = blockIdx.x * 64;
  const int tm0 = blockIdx.y * 64;
  v8f acc0 = {}; v8f acc1 = {};
  gemm_core(A, W, K, tm0, tn0, &Asb[0][0], &Wsb[0][0], acc0, acc1);
  const int tid = threadIdx.x;
  const int wave = tid >> 5;
  const int wm = wave >> 1;
  const int wn = (wave & 1) * 2;
  const int lane = tid & 31;
  const int n = lane & 15;
  const int mb = (lane >> 4) * 8;
  const long row0 = (long)tm0 + wm * 16 + mb;
#pragma unroll
  for (int r = 0; r < 8; ++r) {
    float v0 = acc0[r], v1 = acc1[r];
    if (ACT == 1) { v0 = gelu_exact(v0); v1 = gelu_exact(v1); }
    C[(row0 + r) * N + tn0 + wn * 16 + n]       = (f16)v0;
    C[(row0 + r) * N + tn0 + (wn + 1) * 16 + n] = (f16)v1;
  }
}

// x(strided f32) += A @ W^T
__global__ void __launch_bounds__(256)
gemm_tn_res(const f16* __restrict__ A, const f16* __restrict__ W,
            float* __restrict__ x, int M, int N, int K, Layout L, int Tlen) {
  __shared__ f16 Asb[2][64 * LDC];
  __shared__ f16 Wsb[2][64 * LDC];
  const int tn0 = blockIdx.x * 64;
  const int tm0 = blockIdx.y * 64;
  v8f acc0 = {}; v8f acc1 = {};
  gemm_core(A, W, K, tm0, tn0, &Asb[0][0], &Wsb[0][0], acc0, acc1);
  const int tid = threadIdx.x;
  const int wave = tid >> 5;
  const int wm = wave >> 1;
  const int wn = (wave & 1) * 2;
  const int lane = tid & 31;
  const int n = lane & 15;
  const int mb = (lane >> 4) * 8;
#pragma unroll
  for (int r = 0; r < 8; ++r) {
    int row = tm0 + wm * 16 + mb + r;       // token index
    int seq = row / Tlen, pos = row % Tlen;
    long off = tok_off(L, seq, pos);
    x[off + tn0 + wn * 16 + n]         += acc0[r];
    x[off + (tn0 + (wn + 1) * 16) + n] += acc1[r];
  }
}

// ---------------------------------------------------------------------------
// Attention: one 256-thread block per (sequence, head). TPAD in {256, 64}.
// Q,K row-major [TPAD][64] (head dim zero-padded 48->64); V transposed to
// [64][TPAD]; softmax in registers with wave32 shfl_xor row reductions
// (masks 1..8 stay within the 16-lane halves owning disjoint row sets);
// P converts C-layout -> A-layout through a per-wave LDS strip.
// ---------------------------------------------------------------------------
template <int TPAD>
__global__ void __launch_bounds__(256)
attn_kernel(const f16* __restrict__ Q, const f16* __restrict__ K,
            const f16* __restrict__ V, f16* __restrict__ O,
            int Tlen, float scale) {
  extern __shared__ f16 smem[];
  f16* Qs = smem;                  // TPAD*64
  f16* Ks = Qs + TPAD * 64;        // TPAD*64
  f16* Vt = Ks + TPAD * 64;        // 64*TPAD (transposed)
  f16* Ps = Vt + 64 * TPAD;        // 8 * 16 * TPAD

  const int seq = blockIdx.x / NHEADC;
  const int h   = blockIdx.x % NHEADC;
  const int tid = threadIdx.x;
  const int wave = tid >> 5;
  const int lane = tid & 31;

  for (int idx = tid; idx < TPAD * 8; idx += 256) {
    int row = idx >> 3;
    int c0  = (idx & 7) * 8;
    v8h z = {};
    v8h qv = z, kv = z, vv = z;
    if (row < Tlen && c0 < HDC) {
      long gof = ((long)(seq * Tlen + row)) * DD + h * HDC + c0;
      qv = *(const v8h*)&Q[gof];
      kv = *(const v8h*)&K[gof];
      vv = *(const v8h*)&V[gof];
    }
    *(v8h*)&Qs[row * 64 + c0] = qv;
    *(v8h*)&Ks[row * 64 + c0] = kv;
#pragma unroll
    for (int j = 0; j < 8; ++j) Vt[(c0 + j) * TPAD + row] = vv[j];
  }
  __syncthreads();

  constexpr int NT = TPAD / 16;
  constexpr int NC = TPAD / 32;
  const int n  = lane & 15;
  const int mb = (lane >> 4) * 8;
  f16* Pw = &Ps[wave * 16 * TPAD];

  for (int strip = wave; strip < NT; strip += 8) {
    v16h qa0 = load_a_frag(&Qs[strip * 16 * 64], 64, 0);
    v16h qa1 = load_a_frag(&Qs[strip * 16 * 64], 64, 32);

    v8f acc[NT];
#pragma unroll
    for (int nt = 0; nt < NT; ++nt) {
      v8f a = {};
      v16h b0 = load_b_fragT(&Ks[nt * 16 * 64], 64, 0);
      v16h b1 = load_b_fragT(&Ks[nt * 16 * 64], 64, 32);
      a = wmma16(qa0, b0, a);
      a = wmma16(qa1, b1, a);
      acc[nt] = a;
    }

#pragma unroll
    for (int r = 0; r < 8; ++r) {
      float mx = -3.0e38f;
#pragma unroll
      for (int nt = 0; nt < NT; ++nt) {
        float s = acc[nt][r] * scale;
        if (nt * 16 + n >= Tlen) s = -3.0e38f;
        acc[nt][r] = s;
        mx = fmaxf(mx, s);
      }
#pragma unroll
      for (int m = 8; m >= 1; m >>= 1) mx = fmaxf(mx, __shfl_xor(mx, m));
      float sum = 0.0f;
#pragma unroll
      for (int nt = 0; nt < NT; ++nt) {
        float e = __expf(acc[nt][r] - mx);
        acc[nt][r] = e;
        sum += e;
      }
#pragma unroll
      for (int m = 8; m >= 1; m >>= 1) sum += __shfl_xor(sum, m);
      float inv = 1.0f / sum;
#pragma unroll
      for (int nt = 0; nt < NT; ++nt)
        Pw[(mb + r) * TPAD + nt * 16 + n] = (f16)(acc[nt][r] * inv);
    }

    v8f oa0 = {}, oa1 = {}, oa2 = {};
#pragma unroll
    for (int kc = 0; kc < NC; ++kc) {
      v16h pa = load_a_frag(Pw, TPAD, kc * 32);
      oa0 = wmma16(pa, load_b_fragT(&Vt[0 * 16 * TPAD], TPAD, kc * 32), oa0);
      oa1 = wmma16(pa, load_b_fragT(&Vt[1 * 16 * TPAD], TPAD, kc * 32), oa1);
      oa2 = wmma16(pa, load_b_fragT(&Vt[2 * 16 * TPAD], TPAD, kc * 32), oa2);
    }
#pragma unroll
    for (int r = 0; r < 8; ++r) {
      int row = strip * 16 + mb + r;
      if (row < Tlen) {
        long gof = ((long)(seq * Tlen + row)) * DD + h * HDC;
        O[gof + 0 * 16 + n] = (f16)oa0[r];
        O[gof + 1 * 16 + n] = (f16)oa1[r];
        O[gof + 2 * 16 + n] = (f16)oa2[r];
      }
    }
  }
}

// ---------------------------------------------------------------------------
// Host orchestration
// ---------------------------------------------------------------------------
extern "C" void kernel_launch(void* const* d_in, const int* in_sizes, int n_in,
                              void* d_out, int out_size, void* d_ws, size_t ws_size,
                              hipStream_t stream) {
  const float* xin       = (const float*)d_in[0];
  const float* intra_in  = (const float*)d_in[1];
  const float* intra_op  = (const float*)d_in[2];
  const float* intra_w1  = (const float*)d_in[3];
  const float* intra_w2  = (const float*)d_in[4];
  const float* inter_in  = (const float*)d_in[5];
  const float* inter_op  = (const float*)d_in[6];
  const float* inter_w1  = (const float*)d_in[7];
  const float* inter_w2  = (const float*)d_in[8];
  const float* intra_n1  = (const float*)d_in[9];
  const float* intra_n2  = (const float*)d_in[10];
  const float* inter_n1  = (const float*)d_in[11];
  const float* inter_n2  = (const float*)d_in[12];

  float* x = (float*)d_out;  // working residual stream, canonical (B,NB,T,D)

  const size_t plane  = (size_t)NTOK * DD;
  const size_t welems = (size_t)12 * DD * DD;
  const size_t need = (6 * plane + (size_t)NTOK * FFC + welems) * sizeof(f16);
  if (ws_size < need) return;

  f16* hbuf  = (f16*)d_ws;
  f16* hrbuf = hbuf + plane;
  f16* qb    = hrbuf + plane;
  f16* kb    = qb + plane;
  f16* vb    = kb + plane;
  f16* ob    = vb + plane;
  f16* ffb   = ob + plane;                          // NTOK * FFC
  f16* wip   = ffb + (size_t)NTOK * FFC;            // 3*DD*DD
  f16* wop   = wip + (size_t)3 * DD * DD;           // DD*DD
  f16* w1f   = wop + (size_t)DD * DD;               // FFC*DD
  f16* w2f   = w1f + (size_t)FFC * DD;              // DD*FFC

  (void)hipMemcpyAsync(x, xin, (size_t)out_size * sizeof(float),
                       hipMemcpyDeviceToDevice, stream);

  Layout Lintra;
  Lintra.base = 0; Lintra.seq_div = 0x40000000;
  Lintra.hi_stride = 0;
  Lintra.lo_stride = (long)256 * DD;
  Lintra.pos_stride = DD;

  Layout Linter;
  Linter.base = 0; Linter.seq_div = 256;            // seq = b*256 + t
  Linter.hi_stride = (long)62 * 256 * DD;
  Linter.lo_stride = DD;
  Linter.pos_stride = (long)256 * DD;

  const float scale = 0.14433756729740643f;  // 1/sqrt(48)
  const dim3 gD(DD / 64, NTOK / 64);
  const dim3 gF(FFC / 64, NTOK / 64);

  for (int b = 0; b < NBLK; ++b) {
    for (int ph = 0; ph < 2; ++ph) {
      const bool intra = (ph == 0);
      const float* ip = (intra ? intra_in : inter_in) + (size_t)b * 3 * DD * DD;
      const float* op = (intra ? intra_op : inter_op) + (size_t)b * DD * DD;
      const float* w1 = (intra ? intra_w1 : inter_w1) + (size_t)b * FFC * DD;
      const float* w2 = (intra ? intra_w2 : inter_w2) + (size_t)b * DD * FFC;
      const float* n1 = (intra ? intra_n1 : inter_n1) + (size_t)b * DD;
      const float* n2 = (intra ? intra_n2 : inter_n2) + (size_t)b * DD;
      const Layout L = intra ? Lintra : Linter;
      const int Tlen = intra ? 256 : 62;
      const int Nseq = intra ? 124 : 512;

      cvt_f32_f16<<<512, 256, 0, stream>>>(ip, wip, (long)3 * DD * DD);
      cvt_f32_f16<<<512, 256, 0, stream>>>(op, wop, (long)DD * DD);
      cvt_f32_f16<<<512, 256, 0, stream>>>(w1, w1f, (long)FFC * DD);
      cvt_f32_f16<<<512, 256, 0, stream>>>(w2, w2f, (long)DD * FFC);

      rmsnorm_rope_kernel<<<NTOK, 192, 0, stream>>>(x, n1, hbuf, hrbuf, L, Tlen);

      gemm_tn_f16<0><<<gD, 256, 0, stream>>>(hrbuf, wip,               qb, NTOK, DD, DD);
      gemm_tn_f16<0><<<gD, 256, 0, stream>>>(hrbuf, wip + DD * DD,     kb, NTOK, DD, DD);
      gemm_tn_f16<0><<<gD, 256, 0, stream>>>(hbuf,  wip + 2 * DD * DD, vb, NTOK, DD, DD);

      if (intra) {
        const size_t sh = (size_t)(3 * 256 * 64 + 8 * 16 * 256) * sizeof(f16); // 160 KB
        attn_kernel<256><<<Nseq * NHEADC, 256, sh, stream>>>(qb, kb, vb, ob, Tlen, scale);
      } else {
        const size_t sh = (size_t)(3 * 64 * 64 + 8 * 16 * 64) * sizeof(f16);   // 40 KB
        attn_kernel<64><<<Nseq * NHEADC, 256, sh, stream>>>(qb, kb, vb, ob, Tlen, scale);
      }

      gemm_tn_res<<<gD, 256, 0, stream>>>(ob, wop, x, NTOK, DD, DD, L, Tlen);

      rmsnorm_rope_kernel<<<NTOK, 192, 0, stream>>>(x, n2, hbuf, nullptr, L, Tlen);
      gemm_tn_f16<1><<<gF, 256, 0, stream>>>(hbuf, w1f, ffb, NTOK, FFC, DD);
      gemm_tn_res<<<gD, 256, 0, stream>>>(ffb, w2f, x, NTOK, DD, FFC, L, Tlen);
    }
  }
}
